// TTS_RNN_GCN_5291399708928
// MI455X (gfx1250) — matrix operand order repaired and checked by reading, not compiled, not run on hardware
//
#include <hip/hip_runtime.h>
#include <hip/hip_bf16.h>

// TTS_RNN_GCN for MI455X (gfx1250, wave32, WMMA).
// Pipeline: [P1] c-table precompute, [P2] weight cvt + x-projection collapse,
// [GRU] per-wave 16-row recurrence with v_wmma_f32_16x16x32_f16 (24/step,
//       W_hh B-fragments loop-invariant -> register-hoisted by compiler),
// [GC] h@gc_w^T and h@gc_root_w^T via WMMA, [scatter] edge-normalized
// atomic scatter, [DEC] z@dec_w^T via WMMA + transposed store.
// Transcendentals use v_exp_f32 / v_rcp_f32 (no IEEE div sequences).
// Workspace use: ~15.9 MB.

typedef _Float16 h16_t;
typedef __attribute__((ext_vector_type(16))) _Float16 v16h;
typedef __attribute__((ext_vector_type(8)))  _Float16 v8h;
typedef __attribute__((ext_vector_type(8)))  float    v8f;

#define HID     64
#define NGATE   192
#define NNODES  2000
#define BATCH   8
#define TSTEPS  64
#define NEDGE   32000
#define HORIZON 12
#define ROWS    (BATCH*NNODES)   // 16000

__device__ __forceinline__ v16h make_v16(v8h lo, v8h hi) {
  v16h r;
#pragma unroll
  for (int i = 0; i < 8; ++i) { r[i] = lo[i]; r[i + 8] = hi[i]; }
  return r;
}

__device__ __forceinline__ v8f wmma_f16(v16h a, v16h b, v8f c) {
  return __builtin_amdgcn_wmma_f32_16x16x32_f16(false, a, false, b,
                                                (short)0, c, false, false);
}

// B fragment (32x16 f16) for output column group g..g+15 of a row-major
// [N][64] weight array, K-chunk kc. Lane layout per ISA 7.12.2.
__device__ __forceinline__ v16h load_bfrag(const h16_t* w, int g, int c16, int hi, int kc) {
  const h16_t* br = w + (g + c16) * HID + kc * 32 + hi * 8;
  return make_v16(*(const v8h*)(br), *(const v8h*)(br + 16));
}

// v_rcp_f32 / v_exp_f32 fast paths (avoid IEEE div expansion)
__device__ __forceinline__ float frcp(float x)  { return __builtin_amdgcn_rcpf(x); }
__device__ __forceinline__ float fsig(float x)  { return frcp(1.0f + __expf(-x)); }
__device__ __forceinline__ float ftanh(float x) {
  float e = __expf(2.0f * x);
  return (e - 1.0f) * frcp(e + 1.0f);
}

// ---------------- P1: c[n][g] = b_ih[g] + W_ih[g,:] . (enc_b + node_emb[n,:])
__global__ void k_ctab(const float* __restrict__ wih, const float* __restrict__ bih,
                       const float* __restrict__ encb, const float* __restrict__ nemb,
                       float* __restrict__ ctab) {
  int tid = blockIdx.x * blockDim.x + threadIdx.x;
  if (tid >= NNODES * NGATE) return;
  int n = tid / NGATE, g = tid - n * NGATE;
  float acc = bih[g];
  const float* wr = wih + g * HID;
  const float* er = nemb + n * HID;
#pragma unroll 8
  for (int h = 0; h < HID; ++h) acc += wr[h] * (encb[h] + er[h]);
  ctab[tid] = acc;
}

// ---------------- P2: acomb = W_ih @ enc_w  (192x2); f16 weight copies (dec padded to 32 rows)
__global__ void k_prep(const float* __restrict__ wih, const float* __restrict__ encw,
                       const float* __restrict__ gcw, const float* __restrict__ gcrw,
                       const float* __restrict__ decw,
                       float* __restrict__ acomb, h16_t* __restrict__ gcw16,
                       h16_t* __restrict__ gcr16, h16_t* __restrict__ dec16) {
  int tid = blockIdx.x * blockDim.x + threadIdx.x;
  if (tid < 384) {
    int g = tid >> 1, f = tid & 1;
    float a = 0.f;
#pragma unroll 8
    for (int h = 0; h < HID; ++h) a += wih[g * HID + h] * encw[h * 2 + f];
    acomb[tid] = a;
  } else if (tid < 384 + 4096) {
    int i = tid - 384; gcw16[i] = (h16_t)gcw[i];
  } else if (tid < 384 + 8192) {
    int i = tid - (384 + 4096); gcr16[i] = (h16_t)gcrw[i];
  } else if (tid < 384 + 8192 + 2048) {
    int i = tid - (384 + 8192);
    int r = i >> 6;
    dec16[i] = (r < HORIZON * 2) ? (h16_t)decw[i] : (h16_t)0.0f;
  }
}

// ---------------- GRU: each wave owns 16 rows, loops T=64 steps, 24 WMMA/step.
// Per column-group nt, tiles (nt, nt+4, nt+8) = (r,z,n) are computed and
// consumed immediately so only 3 f32 accumulator tiles are ever live.
__global__ __launch_bounds__(256, 1) void k_gru(
    const float* __restrict__ x, const float* __restrict__ whh,
    const float* __restrict__ bhh, const float* __restrict__ ctab,
    const float* __restrict__ acomb, h16_t* __restrict__ h16out) {
  __shared__ h16_t sWhh[NGATE * HID];   // 24576 B, shared by all waves
  __shared__ h16_t sH[8][16 * HID];     // per-wave f16 hidden state
  __shared__ float sX[8][32];           // per-wave x_t staging (16 rows x 2)

  int tid = threadIdx.x;
  for (int i = tid; i < NGATE * HID; i += 256) sWhh[i] = (h16_t)whh[i];
  __syncthreads();

  int wave = tid >> 5, lane = tid & 31;
  int c16 = lane & 15, hi = lane >> 4;
  int row0 = (blockIdx.x * 8 + wave) * 16;       // 16 consecutive (b,n) rows, never crosses b
  int b  = row0 / NNODES;
  int n0 = row0 - b * NNODES;

  // Per-lane constant preloads. b_hh for r/z gates folds into the c-table;
  // b_hh_n must remain separate (multiplied by the reset gate).
  float crp[4][8], czp[4][8], cnp[4][8];
  float a0r[4], a1r[4], a0z[4], a1z[4], a0n[4], a1n[4], bhn[4];
#pragma unroll
  for (int nt = 0; nt < 4; ++nt) {
    int g = nt * 16 + c16;
    float br = bhh[g], bz = bhh[64 + g];
    bhn[nt] = bhh[128 + g];
#pragma unroll
    for (int v = 0; v < 8; ++v) {
      int rl = v + 8 * hi;
      const float* crow = ctab + (size_t)(row0 + rl) * NGATE;
      crp[nt][v] = crow[g] + br;
      czp[nt][v] = crow[64 + g] + bz;
      cnp[nt][v] = crow[128 + g];
    }
    a0r[nt] = acomb[2 * g];           a1r[nt] = acomb[2 * g + 1];
    a0z[nt] = acomb[2 * (64 + g)];    a1z[nt] = acomb[2 * (64 + g) + 1];
    a0n[nt] = acomb[2 * (128 + g)];   a1n[nt] = acomb[2 * (128 + g) + 1];
  }

  float hreg[4][8];
  h16_t* hb = &sH[wave][0];
#pragma unroll
  for (int nt = 0; nt < 4; ++nt)
#pragma unroll
    for (int v = 0; v < 8; ++v) {
      hreg[nt][v] = 0.f;
      hb[(v + 8 * hi) * HID + nt * 16 + c16] = (h16_t)0.f;
    }

  for (int t = 0; t < TSTEPS; ++t) {
    size_t xoff = (((size_t)b * TSTEPS + t) * NNODES + n0) * 2;
    sX[wave][lane] = x[xoff + lane];                // 32 floats = 16 rows x 2 feats
    float xv0[8], xv1[8];
#pragma unroll
    for (int v = 0; v < 8; ++v) {
      int rl = v + 8 * hi;
      xv0[v] = sX[wave][rl * 2]; xv1[v] = sX[wave][rl * 2 + 1];
    }

    // A fragments (h as 16x32 f16, two K-chunks) — ISA 7.12.2 layout
    const h16_t* ar = hb + c16 * HID;
    v16h a0 = make_v16(*(const v8h*)(ar + hi * 8),       *(const v8h*)(ar + 16 + hi * 8));
    v16h a1 = make_v16(*(const v8h*)(ar + 32 + hi * 8),  *(const v8h*)(ar + 48 + hi * 8));

#pragma unroll
    for (int nt = 0; nt < 4; ++nt) {
      int gr0 = nt * 16, gz0 = 64 + nt * 16, gn0 = 128 + nt * 16;
      v8f accR = {}, accZ = {}, accN = {};
      accR = wmma_f16(a0, load_bfrag(sWhh, gr0, c16, hi, 0), accR);
      accZ = wmma_f16(a0, load_bfrag(sWhh, gz0, c16, hi, 0), accZ);
      accN = wmma_f16(a0, load_bfrag(sWhh, gn0, c16, hi, 0), accN);
      accR = wmma_f16(a1, load_bfrag(sWhh, gr0, c16, hi, 1), accR);
      accZ = wmma_f16(a1, load_bfrag(sWhh, gz0, c16, hi, 1), accZ);
      accN = wmma_f16(a1, load_bfrag(sWhh, gn0, c16, hi, 1), accN);

#pragma unroll
      for (int v = 0; v < 8; ++v) {
        float gr  = fmaf(xv0[v], a0r[nt], fmaf(xv1[v], a1r[nt], crp[nt][v])) + accR[v];
        float gz  = fmaf(xv0[v], a0z[nt], fmaf(xv1[v], a1z[nt], czp[nt][v])) + accZ[v];
        float gin = fmaf(xv0[v], a0n[nt], fmaf(xv1[v], a1n[nt], cnp[nt][v]));
        float hn  = accN[v] + bhn[nt];
        float r  = fsig(gr);
        float z  = fsig(gz);
        float ct = ftanh(gin + r * hn);
        float h  = (1.0f - z) * ct + z * hreg[nt][v];
        hreg[nt][v] = h;
        hb[(v + 8 * hi) * HID + nt * 16 + c16] = (h16_t)h;
      }
    }
  }

#pragma unroll
  for (int nt = 0; nt < 4; ++nt)
#pragma unroll
    for (int v = 0; v < 8; ++v) {
      int rg = row0 + v + 8 * hi;
      h16out[(size_t)rg * HID + nt * 16 + c16] = (h16_t)hreg[nt][v];
    }
}

// ---------------- GC: xlin = h@gc_w^T, root = h@gc_root_w^T (WMMA, A from global h16)
__global__ __launch_bounds__(256, 1) void k_gc_lin(
    const h16_t* __restrict__ h16, const h16_t* __restrict__ gcw16,
    const h16_t* __restrict__ gcr16, float* __restrict__ xlin,
    float* __restrict__ root) {
  int tid = threadIdx.x;
  int wave = tid >> 5, lane = tid & 31, c16 = lane & 15, hi = lane >> 4;
  int row0 = (blockIdx.x * 8 + wave) * 16;

  const h16_t* hr = h16 + (size_t)(row0 + c16) * HID;
  v16h a0 = make_v16(*(const v8h*)(hr + hi * 8),      *(const v8h*)(hr + 16 + hi * 8));
  v16h a1 = make_v16(*(const v8h*)(hr + 32 + hi * 8), *(const v8h*)(hr + 48 + hi * 8));

#pragma unroll
  for (int nt = 0; nt < 4; ++nt) {
    int g = nt * 16;
    v8f cw = {}, cc = {};
    cw = wmma_f16(a0, load_bfrag(gcw16, g, c16, hi, 0), cw);
    cw = wmma_f16(a1, load_bfrag(gcw16, g, c16, hi, 1), cw);
    cc = wmma_f16(a0, load_bfrag(gcr16, g, c16, hi, 0), cc);
    cc = wmma_f16(a1, load_bfrag(gcr16, g, c16, hi, 1), cc);
#pragma unroll
    for (int v = 0; v < 8; ++v) {
      size_t idx = (size_t)(row0 + v + 8 * hi) * HID + nt * 16 + c16;
      xlin[idx] = cw[v];
      root[idx] = cc[v];
    }
  }
}

// ---------------- zero agg + deg
__global__ void k_zero(float* __restrict__ agg, float* __restrict__ deg) {
  int tid = blockIdx.x * blockDim.x + threadIdx.x;
  if (tid < ROWS * HID) agg[tid] = 0.f;
  else if (tid < ROWS * HID + NNODES) deg[tid - ROWS * HID] = 0.f;
}

// ---------------- deg[dst] += w
__global__ void k_deg(const int* __restrict__ ei, const float* __restrict__ ew,
                      float* __restrict__ deg) {
  int e = blockIdx.x * blockDim.x + threadIdx.x;
  if (e >= NEDGE) return;
  unsafeAtomicAdd(&deg[ei[NEDGE + e]], ew[e]);
}

// ---------------- edge scatter: agg[b,dst,h] += xlin[b,src,h] * w/deg[dst]
__global__ void k_scatter(const int* __restrict__ ei, const float* __restrict__ ew,
                          const float* __restrict__ deg, const float* __restrict__ xlin,
                          float* __restrict__ agg) {
  int gid = blockIdx.x * blockDim.x + threadIdx.x;
  if (gid >= NEDGE * HID) return;
  int e = gid >> 6, hc = gid & 63;
  int s = ei[e], d = ei[NEDGE + e];
  float wn = ew[e] * __builtin_amdgcn_rcpf(fmaxf(deg[d], 1e-12f));
#pragma unroll
  for (int b = 0; b < BATCH; ++b) {
    float v = xlin[((size_t)(b * NNODES + s)) * HID + hc] * wn;
    unsafeAtomicAdd(&agg[((size_t)(b * NNODES + d)) * HID + hc], v);
  }
}

// ---------------- decoder: z = agg+root+gc_b ; out = z@dec_w^T + dec_b ; transposed store
__global__ __launch_bounds__(256, 1) void k_dec(
    const float* __restrict__ agg, const float* __restrict__ root,
    const float* __restrict__ gcb, const h16_t* __restrict__ dec16,
    const float* __restrict__ decb, float* __restrict__ out) {
  __shared__ h16_t sZ[8][16 * HID];
  int tid = threadIdx.x;
  int wave = tid >> 5, lane = tid & 31, c16 = lane & 15, hi = lane >> 4;
  int row0 = (blockIdx.x * 8 + wave) * 16;
  h16_t* zb = &sZ[wave][0];

#pragma unroll
  for (int nt = 0; nt < 4; ++nt)
#pragma unroll
    for (int v = 0; v < 8; ++v) {
      int rl = v + 8 * hi, col = nt * 16 + c16;
      size_t idx = (size_t)(row0 + rl) * HID + col;
      zb[rl * HID + col] = (h16_t)(agg[idx] + root[idx] + gcb[col]);
    }
  // same-wave LDS store->load is in-order in the DS pipe

  const h16_t* ar = zb + c16 * HID;
  v16h a0 = make_v16(*(const v8h*)(ar + hi * 8),      *(const v8h*)(ar + 16 + hi * 8));
  v16h a1 = make_v16(*(const v8h*)(ar + 32 + hi * 8), *(const v8h*)(ar + 48 + hi * 8));

  v8f acc[2];
#pragma unroll
  for (int nt = 0; nt < 2; ++nt) {
    v8f c = {};
    c = wmma_f16(a0, load_bfrag(dec16, nt * 16, c16, hi, 0), c);  // rows 24..31 zero-padded
    c = wmma_f16(a1, load_bfrag(dec16, nt * 16, c16, hi, 1), c);
    acc[nt] = c;
  }

#pragma unroll
  for (int nt = 0; nt < 2; ++nt) {
    int col = nt * 16 + c16;
    if (col < HORIZON * 2) {
      float db = decb[col];
      int tt = col >> 1, f = col & 1;
#pragma unroll
      for (int v = 0; v < 8; ++v) {
        int rg = row0 + v + 8 * hi;
        int bb = rg / NNODES, nn = rg - bb * NNODES;
        out[(((size_t)bb * HORIZON + tt) * NNODES + nn) * 2 + f] = acc[nt][v] + db;
      }
    }
  }
}

// ---------------- host launcher
extern "C" void kernel_launch(void* const* d_in, const int* in_sizes, int n_in,
                              void* d_out, int out_size, void* d_ws, size_t ws_size,
                              hipStream_t stream) {
  const float* x    = (const float*)d_in[0];
  const int*   ei   = (const int*)d_in[1];
  const float* ew   = (const float*)d_in[2];
  const float* encw = (const float*)d_in[3];
  const float* encb = (const float*)d_in[4];
  const float* nemb = (const float*)d_in[5];
  const float* wih  = (const float*)d_in[6];
  const float* whh  = (const float*)d_in[7];
  const float* bih  = (const float*)d_in[8];
  const float* bhh  = (const float*)d_in[9];
  const float* gcw  = (const float*)d_in[10];
  const float* gcrw = (const float*)d_in[11];
  const float* gcb  = (const float*)d_in[12];
  const float* decw = (const float*)d_in[13];
  const float* decb = (const float*)d_in[14];
  float* out = (float*)d_out;
  char*  ws  = (char*)d_ws;

  // workspace layout (all offsets 16B-aligned), ~15.9 MB total
  float*  ws_ctab  = (float*)(ws + 0);          // 2000*192*4   = 1,536,000
  float*  ws_acomb = (float*)(ws + 1536000);    // 384*4        = 1,536
  h16_t*  ws_h16   = (h16_t*)(ws + 1537536);    // 16000*64*2   = 2,048,000
  float*  ws_xlin  = (float*)(ws + 3585536);    // 16000*64*4   = 4,096,000
  float*  ws_root  = (float*)(ws + 7681536);    // 4,096,000
  float*  ws_agg   = (float*)(ws + 11777536);   // 4,096,000
  float*  ws_deg   = (float*)(ws + 15873536);   // 2000*4 = 8,000
  h16_t*  ws_gcw16 = (h16_t*)(ws + 15881536);   // 64*64*2 = 8,192
  h16_t*  ws_gcr16 = (h16_t*)(ws + 15889728);   // 8,192
  h16_t*  ws_dec16 = (h16_t*)(ws + 15897920);   // 32*64*2 = 4,096

  k_ctab<<<(NNODES * NGATE + 255) / 256, 256, 0, stream>>>(wih, bih, encb, nemb, ws_ctab);
  k_prep<<<(384 + 8192 + 2048 + 255) / 256, 256, 0, stream>>>(
      wih, encw, gcw, gcrw, decw, ws_acomb, ws_gcw16, ws_gcr16, ws_dec16);
  k_zero<<<(ROWS * HID + NNODES + 255) / 256, 256, 0, stream>>>(ws_agg, ws_deg);

  k_gru<<<ROWS / 128, 256, 0, stream>>>(x, whh, bhh, ws_ctab, ws_acomb, ws_h16);

  k_gc_lin<<<ROWS / 128, 256, 0, stream>>>(ws_h16, ws_gcw16, ws_gcr16, ws_xlin, ws_root);

  k_deg<<<(NEDGE + 255) / 256, 256, 0, stream>>>(ei, ew, ws_deg);
  k_scatter<<<(NEDGE * HID) / 256, 256, 0, stream>>>(ei, ew, ws_deg, ws_xlin, ws_agg);

  k_dec<<<ROWS / 128, 256, 0, stream>>>(ws_agg, ws_root, gcb, ws_dec16, decb, out);
}
